// KGPrompt_48722109006434
// MI455X (gfx1250) — matrix operand order repaired and checked by reading, not compiled, not run on hardware
//
#include <hip/hip_runtime.h>
#include <hip/hip_bf16.h>
#include <math.h>

// ---------------------------------------------------------------------------
// KGPrompt fused kernel set for MI455X (gfx1250, wave32, WMMA).
//
// Workspace layout (assumed ws_size >= ~108 MB):
//   [0)                      E_bf16 : VOCAB*H  __bf16   (102.9 MB)
//   [VOCAB*H)                Q_bf16 : RQ*H     __bf16   (  4.9 MB)
// ---------------------------------------------------------------------------

typedef __attribute__((ext_vector_type(16))) __bf16 v16bf;
typedef __attribute__((ext_vector_type(8)))  float  v8f;
typedef __attribute__((ext_vector_type(4)))  int    i32x4;

#define H      1024
#define VOCABN 50257
#define RQ     2400      // 16 * 150 query rows
#define NB     16
#define TQ     150
#define CTXL   256
#define TOUT   406       // 150 + 256
#define VT     64        // vocab tile (chunk of E rows per iteration)
#define MT     32        // query rows per workgroup

// CDNA5 async memory->LDS path (ASYNCcnt), if the toolchain exposes it.
#if __has_builtin(__builtin_amdgcn_global_load_async_to_lds_b128)
#define KG_ASYNC_LDS 1
#endif

__device__ __forceinline__ v8f wmma_bf16(v16bf a, v16bf b, v8f c) {
    return __builtin_amdgcn_wmma_f32_16x16x32_bf16(false, a, false, b,
                                                   (short)0, c, false, false);
}

// A-operand (16x32 bf16): lane hi=0 -> K base..base+15 contiguous
//                         lane hi=1 -> (caller passes base+16) same pattern.
__device__ __forceinline__ v16bf ldA_f32(const float* p) {
    v16bf r;
#pragma unroll
    for (int i = 0; i < 16; ++i) r[i] = (__bf16)p[i];
    return r;
}
__device__ __forceinline__ v16bf ldA_bf(const __bf16* p) {
    v16bf r;
#pragma unroll
    for (int i = 0; i < 16; ++i) r[i] = p[i];
    return r;
}
// B-operand (32x16 bf16): lane hi=0 -> K 0..15 contiguous; hi=1 -> K 16..31.
__device__ __forceinline__ v16bf ldB_f32(const float* p) {
    v16bf r;
#pragma unroll
    for (int i = 0; i < 16; ++i) r[i] = (__bf16)p[i];
    return r;
}
__device__ __forceinline__ v16bf ldB_bf(const __bf16* p) {
    v16bf r;
#pragma unroll
    for (int i = 0; i < 16; ++i) r[i] = p[i];
    return r;
}

// ---------------------------------------------------------------------------
// K0: convert word_embeddings fp32 -> bf16 (makes E L2-resident: 103 MB < 192 MB)
// ---------------------------------------------------------------------------
__global__ __launch_bounds__(256) void cvt_e_bf16(const float* __restrict__ e,
                                                  __bf16* __restrict__ o,
                                                  size_t n4) {
    size_t stride = (size_t)gridDim.x * 256;
    for (size_t v = (size_t)blockIdx.x * 256 + threadIdx.x; v < n4; v += stride) {
        float4 f = ((const float4*)e)[v];
        __bf16* d = o + v * 4;
        d[0] = (__bf16)f.x; d[1] = (__bf16)f.y; d[2] = (__bf16)f.z; d[3] = (__bf16)f.w;
    }
}

// ---------------------------------------------------------------------------
// K1: Q = prompt @ W^T   (2400x1024x1024, bf16 WMMA, fp32 acc, bf16 out)
// grid = 75, block = 256 (8 waves).  Wave w owns 128 output columns.
// ---------------------------------------------------------------------------
__global__ __launch_bounds__(256, 1) void qgemm(const float* __restrict__ P,
                                                const float* __restrict__ W,
                                                __bf16* __restrict__ Q) {
    const int tid = threadIdx.x;
    const int w = tid >> 5, lane = tid & 31, lm = lane & 15, hi = lane >> 4;
    const int row0 = blockIdx.x * MT;
    const int nbase = w * 128;

    v8f acc[2][8];
#pragma unroll
    for (int a = 0; a < 2; ++a)
#pragma unroll
        for (int b = 0; b < 8; ++b)
#pragma unroll
            for (int e = 0; e < 8; ++e) acc[a][b][e] = 0.0f;

    const float* prow0 = P + (size_t)(row0 + lm) * H + hi * 16;
    const float* prow1 = prow0 + (size_t)16 * H;

    for (int k0 = 0; k0 < H; k0 += 32) {
        v16bf a0 = ldA_f32(prow0 + k0);
        v16bf a1 = ldA_f32(prow1 + k0);
#pragma unroll
        for (int nt = 0; nt < 8; ++nt) {
            v16bf b = ldB_f32(W + (size_t)(nbase + nt * 16 + lm) * H + hi * 16 + k0);
            acc[0][nt] = wmma_bf16(a0, b, acc[0][nt]);
            acc[1][nt] = wmma_bf16(a1, b, acc[1][nt]);
        }
    }
    // C/D layout: VGPR j -> row j + 8*hi, lane lm -> column
#pragma unroll
    for (int mt = 0; mt < 2; ++mt)
#pragma unroll
        for (int j = 0; j < 8; ++j) {
            __bf16* dst = Q + (size_t)(row0 + mt * 16 + hi * 8 + j) * H + nbase + lm;
#pragma unroll
            for (int nt = 0; nt < 8; ++nt) dst[nt * 16] = (__bf16)acc[mt][nt][j];
        }
}

// ---------------------------------------------------------------------------
// K2: fused  S = Q@E^T/1024 ; P = online-softmax(S) ; out += P@E
// grid = 75 (32 rows each), block = 256.
// LDS: E chunk 64x1024 bf16 (128KB) + P 32x64 bf16 (4KB) + stats (~1.4KB)
//      => ~136KB dynamic LDS: two workgroups still fit a 320KB WGP.
// acc (32x1024 fp32) lives in VGPRs: wave w owns cols [w*128, w*128+128).
// ---------------------------------------------------------------------------
__global__ __launch_bounds__(256, 1) void fused_softmax_gemm(
        const __bf16* __restrict__ wsQ, const __bf16* __restrict__ wsE,
        float* __restrict__ out) {
    extern __shared__ char smem[];
    __bf16* ldsE = (__bf16*)smem;                    // VT*H
    __bf16* ldsP = ldsE + (size_t)VT * H;            // MT*VT
    float*  m_run = (float*)(ldsP + (size_t)MT * VT);
    float*  l_run = m_run + MT;
    float*  alf   = l_run + MT;
    float*  cmax  = alf + MT;                        // 4*MT
    float*  csum  = cmax + 4 * MT;                   // 4*MT

    const int tid = threadIdx.x;
    const int w = tid >> 5, lane = tid & 31, lm = lane & 15, hi = lane >> 4;
    const int row0 = blockIdx.x * MT;
    const int s_mt = w & 1;          // score pass: which 16-row tile
    const int s_nt = w >> 1;         // score pass: which 16-vocab tile (0..3)
    const int nbase = w * 128;       // mapped pass: owned hidden columns

    if (tid < MT) { m_run[tid] = -3.0e38f; l_run[tid] = 0.0f; }

    v8f acc[2][8];
#pragma unroll
    for (int a = 0; a < 2; ++a)
#pragma unroll
        for (int b = 0; b < 8; ++b)
#pragma unroll
            for (int e = 0; e < 8; ++e) acc[a][b][e] = 0.0f;

    const __bf16* qrow = wsQ + (size_t)(row0 + s_mt * 16 + lm) * H + hi * 16;

    for (int v0 = 0; v0 < VOCABN; v0 += VT) {
        __syncthreads();   // previous mapped pass done before restaging E
        {   // stage E chunk (rows are contiguous in wsE) -> LDS, zero OOB tail
            const uint4* src = (const uint4*)(wsE + (size_t)v0 * H);
            uint4* dst = (uint4*)ldsE;
            const int nvec = VT * H / 8;                                  // 8192
            const int valid = ((VOCABN - v0) >= VT) ? nvec : ((VOCABN - v0) * H) / 8;
            uint4 z; z.x = z.y = z.z = z.w = 0u;
#ifdef KG_ASYNC_LDS
            // CDNA5 async DMA: memory -> LDS without touching VGPRs (ASYNCcnt)
            for (int i = tid; i < valid; i += 256)
                __builtin_amdgcn_global_load_async_to_lds_b128(
                    (__attribute__((address_space(1))) i32x4*)(src + i),
                    (__attribute__((address_space(3))) i32x4*)(dst + i), 0, 0);
            for (int i = valid + tid; i < nvec; i += 256) dst[i] = z;
#if __has_builtin(__builtin_amdgcn_s_wait_asynccnt)
            __builtin_amdgcn_s_wait_asynccnt(0);
#else
            asm volatile("s_wait_asynccnt 0x0" ::: "memory");
#endif
#else
            for (int i = tid; i < nvec; i += 256) dst[i] = (i < valid) ? src[i] : z;
#endif
        }
        // prefetch next chunk toward L2 while we compute on this one
        if (v0 + VT < VOCABN)
            __builtin_prefetch(wsE + (size_t)(v0 + VT) * H + (size_t)tid * 512, 0, 0);
        __syncthreads();

        // ---- scores: each wave one 16x16 tile of S[32, VT], K = 1024 ----
        v8f s;
#pragma unroll
        for (int e = 0; e < 8; ++e) s[e] = 0.0f;
        const __bf16* erow = ldsE + (size_t)(s_nt * 16 + lm) * H + hi * 16;
#pragma unroll 4
        for (int k0 = 0; k0 < H; k0 += 32)
            s = wmma_bf16(ldA_bf(qrow + k0), ldB_bf(erow + k0), s);

        const int col = v0 + s_nt * 16 + lm;
        float sv[8];
#pragma unroll
        for (int j = 0; j < 8; ++j)
            sv[j] = (col < VOCABN) ? s[j] * (1.0f / 1024.0f) : -3.0e38f;

        // ---- per-row chunk max (16-lane xor reduction, rows = j + 8*hi) ----
#pragma unroll
        for (int j = 0; j < 8; ++j) {
            float r = sv[j];
            r = fmaxf(r, __shfl_xor(r, 1));
            r = fmaxf(r, __shfl_xor(r, 2));
            r = fmaxf(r, __shfl_xor(r, 4));
            r = fmaxf(r, __shfl_xor(r, 8));
            if (lm == 0) cmax[s_nt * MT + s_mt * 16 + hi * 8 + j] = r;
        }
        __syncthreads();
        if (tid < MT) {
            float mc = fmaxf(fmaxf(cmax[tid], cmax[MT + tid]),
                             fmaxf(cmax[2 * MT + tid], cmax[3 * MT + tid]));
            float mold = m_run[tid];
            float mnew = fmaxf(mold, mc);
            alf[tid] = __expf(mold - mnew);   // 0 on first chunk
            m_run[tid] = mnew;
        }
        __syncthreads();

        // ---- exp, write P (bf16) to LDS, per-row partial sums ----
#pragma unroll
        for (int j = 0; j < 8; ++j) {
            const int rloc = s_mt * 16 + hi * 8 + j;
            float p = __expf(sv[j] - m_run[rloc]);   // masked cols -> exp(-huge)=0
            ldsP[(size_t)rloc * VT + s_nt * 16 + lm] = (__bf16)p;
            float rs = p;
            rs += __shfl_xor(rs, 1);
            rs += __shfl_xor(rs, 2);
            rs += __shfl_xor(rs, 4);
            rs += __shfl_xor(rs, 8);
            if (lm == 0) csum[s_nt * MT + rloc] = rs;
        }
        __syncthreads();
        if (tid < MT) {
            float s4 = csum[tid] + csum[MT + tid] + csum[2 * MT + tid] + csum[3 * MT + tid];
            l_run[tid] = l_run[tid] * alf[tid] + s4;
        }

        // ---- rescale running accumulator by alpha[row] ----
#pragma unroll
        for (int mt = 0; mt < 2; ++mt)
#pragma unroll
            for (int j = 0; j < 8; ++j) {
                float a_ = alf[mt * 16 + hi * 8 + j];
#pragma unroll
                for (int nt = 0; nt < 8; ++nt) acc[mt][nt][j] *= a_;
            }

        // ---- mapped: acc[32, 1024] += P[32,VT] @ Echunk[VT,1024] ----
#pragma unroll
        for (int ks = 0; ks < VT / 32; ++ks) {
            const int kb = ks * 32;
            v16bf aP[2];
#pragma unroll
            for (int mt = 0; mt < 2; ++mt)
                aP[mt] = ldA_bf(ldsP + (size_t)(mt * 16 + lm) * VT + kb + hi * 16);
#pragma unroll
            for (int nt = 0; nt < 8; ++nt) {
                // B[k,n] = E[v0+kb+hi*16+k][col]: column gather from row-major LDS
                const __bf16* bp = ldsE + (size_t)(kb + hi * 16) * H + nbase + nt * 16 + lm;
                v16bf b;
#pragma unroll
                for (int i = 0; i < 16; ++i) b[i] = bp[(size_t)i * H];
                acc[0][nt] = wmma_bf16(aP[0], b, acc[0][nt]);
                acc[1][nt] = wmma_bf16(aP[1], b, acc[1][nt]);
            }
        }
    }

    __syncthreads();
    // ---- normalize by l and scatter into out_embeds prompt rows ----
#pragma unroll
    for (int mt = 0; mt < 2; ++mt)
#pragma unroll
        for (int j = 0; j < 8; ++j) {
            const int rloc = mt * 16 + hi * 8 + j;
            const int rg = row0 + rloc;
            const int b_ = rg / TQ, t_ = rg % TQ;
            const float inv = 1.0f / l_run[rloc];
            float* dst = out + ((size_t)b_ * TOUT + t_) * H + nbase + lm;
#pragma unroll
            for (int nt = 0; nt < 8; ++nt) dst[nt * 16] = acc[mt][nt][j] * inv;
        }
}

// ---------------------------------------------------------------------------
// K3: copy context embeddings into out rows [150,406) and write the mask.
// ---------------------------------------------------------------------------
__global__ __launch_bounds__(256) void epilogue(const float* __restrict__ ctx,
                                                const float* __restrict__ attn,
                                                float* __restrict__ out) {
    const size_t stride = (size_t)gridDim.x * 256;
    const size_t i = (size_t)blockIdx.x * 256 + threadIdx.x;
    // 16*256*1024 / 4 = 1,048,576 float4 copies
    for (size_t v = i; v < (size_t)NB * CTXL * H / 4; v += stride) {
        const size_t e = v * 4;
        const size_t b = e >> 18;           // / (256*1024)
        const size_t rem = e & 262143;      // % (256*1024)
        *((float4*)(out + ((size_t)b * TOUT + TQ) * H + rem)) = ((const float4*)ctx)[v];
    }
    float* mask = out + (size_t)NB * TOUT * H;
    for (size_t v = i; v < (size_t)NB * TOUT; v += stride) {
        const size_t b = v / TOUT, t = v % TOUT;
        mask[v] = (t < TQ) ? 1.0f : attn[b * CTXL + (t - TQ)];
    }
}

// ---------------------------------------------------------------------------
extern "C" void kernel_launch(void* const* d_in, const int* in_sizes, int n_in,
                              void* d_out, int out_size, void* d_ws, size_t ws_size,
                              hipStream_t stream) {
    const float* token_embeds = (const float*)d_in[0];   // [2400, 1024] (flat prompt)
    const float* word_embed   = (const float*)d_in[1];   // [50257, 1024]
    const float* ctx          = (const float*)d_in[2];   // [16, 256, 1024]
    const float* attn         = (const float*)d_in[3];   // [16, 256]
    const float* W_cross      = (const float*)d_in[4];   // [1024, 1024]
    float* out = (float*)d_out;

    __bf16* wsE = (__bf16*)d_ws;
    __bf16* wsQ = wsE + (size_t)VOCABN * H;

    (void)in_sizes; (void)n_in; (void)out_size; (void)ws_size;

    // K0: E -> bf16 (L2-resident table for both big GEMM passes)
    cvt_e_bf16<<<8192, 256, 0, stream>>>(word_embed, wsE, (size_t)VOCABN * H / 4);

    // K1: Q = prompt @ W^T  (bf16 out)
    qgemm<<<RQ / MT, 256, 0, stream>>>(token_embeds, W_cross, wsQ);

    // K2: fused scores + online softmax + P@E, writes prompt rows of out
    const size_t smem = ((size_t)VT * H + (size_t)MT * VT) * sizeof(__bf16)
                      + (size_t)(3 * MT + 8 * MT) * sizeof(float);
    (void)hipFuncSetAttribute((const void*)fused_softmax_gemm,
                              hipFuncAttributeMaxDynamicSharedMemorySize, (int)smem);
    fused_softmax_gemm<<<RQ / MT, 256, smem, stream>>>(wsQ, wsE, out);

    // K3: context copy + mask
    epilogue<<<4096, 256, 0, stream>>>(ctx, attn, out);
}
